// JambaMambaMixer_27736898797981
// MI455X (gfx1250) — matrix operand order repaired and checked by reading, not compiled, not run on hardware
//
#include <hip/hip_runtime.h>
#include <hip/hip_bf16.h>

// ---------------- problem constants (match reference) ----------------
#define HDIM 2048
#define DI   4096
#define NS   16
#define KW   4
#define RK   128
#define BSZ  2
#define LSEQ 2048
#define BL   (BSZ * LSEQ)      // 4096 token rows
#define PROJW (2 * DI)         // 8192
#define SSMW  (RK + 2 * NS)    // 160

typedef __attribute__((ext_vector_type(16))) __bf16          v16bf;
typedef __attribute__((ext_vector_type(8)))  float           v8f;
typedef __attribute__((ext_vector_type(16))) unsigned short  u16x16;
typedef __attribute__((ext_vector_type(8)))  unsigned short  u16x8;

// fp32 -> bf16 round-to-nearest-even
__device__ __forceinline__ unsigned short f32_to_bf16(float f) {
    unsigned int u = __builtin_bit_cast(unsigned int, f);
    unsigned int r = u + 0x7FFFu + ((u >> 16) & 1u);
    return (unsigned short)(r >> 16);
}

__device__ __forceinline__ float silu_f(float x) {
    return x / (1.0f + __expf(-x));
}

// gfx1250 async global->LDS copy (ASYNCcnt-tracked), per ISA ch.10 / 08_async_tensor.
__device__ __forceinline__ void async_load_b128(unsigned lds_off, const unsigned short* g) {
    asm volatile("global_load_async_to_lds_b128 %0, %1, off"
                 :: "v"(lds_off), "v"(g) : "memory");
}
__device__ __forceinline__ void wait_asynccnt0() {
    asm volatile("s_wait_asynccnt 0x0" ::: "memory");
}

// ---------------- elementwise fp32 -> bf16 cast ----------------
__global__ void cast_f32_to_bf16_kernel(const float* __restrict__ in,
                                        unsigned short* __restrict__ out,
                                        size_t n) {
    size_t i = (size_t)blockIdx.x * blockDim.x + threadIdx.x;
    if (i < n) out[i] = f32_to_bf16(in[i]);
}

// ===========================================================================
// Tiled bf16 WMMA GEMM: C(MxN,f32) = A(MxK,bf16) * B(KxN,bf16)
// Block: 256 thr (8 waves, 2x4), 128x128 output tile, K stepped by 32.
//  - double-buffered LDS (ping-pong): async A copy + global B load of tile
//    k+32 issued before computing tile k; one barrier per K step
//  - A tile (128x32) staged via global_load_async_to_lds_b128 (ASYNCcnt)
//  - B tile (32x128) loaded coalesced to regs, scattered transposed to LDS
//  - each wave: 4x2 sub-tiles => 8 v_wmma_f32_16x16x32_bf16 per K step
// Requires: M%128==0, N%128==0, K%32==0.
// ===========================================================================
#define BLK_M 128
#define BLK_N 128
#define KSTEP 32
#define BPAD  40   // padded k-stride of transposed B tile (16B-aligned rows, conflict-free scatter)

__global__ __launch_bounds__(256)
void wmma_gemm_bf16_tiled_kernel(const unsigned short* __restrict__ A,
                                 const unsigned short* __restrict__ B,
                                 float* __restrict__ C,
                                 int M, int N, int K) {
    __shared__ alignas(32) unsigned short Atile[2][BLK_M][KSTEP];  // 16 KB
    __shared__ alignas(32) unsigned short Btile[2][BLK_N][BPAD];   // 20 KB

    const int tid  = threadIdx.x;
    const int lane = tid & 31;
    const int wv   = tid >> 5;          // 0..7
    const int wm   = wv >> 2;           // 0..1 (M dir, 64 rows each)
    const int wn   = wv & 3;            // 0..3 (N dir, 32 cols each)
    const int l16  = lane & 15;
    const int lh   = lane >> 4;         // lane half

    const int nblk = N / BLK_N;
    const int bm   = blockIdx.x / nblk;
    const int bn   = blockIdx.x - bm * nblk;
    const int m0   = bm * BLK_M;
    const int n0   = bn * BLK_N;

    // A staging: one b128 (8 bf16) per thread; 2 threads per row
    const int ar = tid >> 1;
    const int ac = (tid & 1) * 16;
    const unsigned a_lds0 = (unsigned)(uintptr_t)&Atile[0][ar][ac];
    const unsigned a_lds1 = (unsigned)(uintptr_t)&Atile[1][ar][ac];
    const unsigned short* Ag = A + (size_t)(m0 + ar) * K + ac;

    // B staging: 16 bf16 per thread (two b128), 8 threads per k-row
    const int br = tid >> 3;            // k-row 0..31
    const int bc = (tid & 7) * 16;      // col segment base
    const unsigned short* Bg = B + (size_t)br * N + n0 + bc;

    v8f acc[4][2] = {};                 // [mt][nt]

    // ---- prologue: stage tile k=0 into buffer 0
    async_load_b128(a_lds0, Ag);
    {
        const u16x8 b0 = *(const u16x8*)(Bg);
        const u16x8 b1 = *(const u16x8*)(Bg + 8);
#pragma unroll
        for (int j = 0; j < 8; ++j) {
            Btile[0][bc + j][br]     = b0[j];
            Btile[0][bc + 8 + j][br] = b1[j];
        }
    }
    wait_asynccnt0();
    __syncthreads();

    for (int k0 = 0; k0 < K; k0 += KSTEP) {
        const int cur = (k0 / KSTEP) & 1;
        const int nxt = cur ^ 1;
        const bool has_next = (k0 + KSTEP) < K;   // uniform branch

        // ---- issue next-tile staging early: async A DMA + global B -> regs
        u16x8 nb0 = {}, nb1 = {};
        if (has_next) {
            const int kn = k0 + KSTEP;
            async_load_b128(cur ? a_lds0 : a_lds1, Ag + kn);
            const unsigned short* Bn = Bg + (size_t)kn * N;
            nb0 = *(const u16x8*)(Bn);
            nb1 = *(const u16x8*)(Bn + 8);
        }

        // ---- compute current tile: 4 (M) x 2 (N) sub-tiles per wave
#pragma unroll
        for (int mt = 0; mt < 4; ++mt) {
            const int arow = wm * 64 + mt * 16 + l16;
            // A fragment: av[0..7] = K lh*8..+7 ; av[8..15] = K 16+lh*8..+7
            const u16x8 alo = *(const u16x8*)&Atile[cur][arow][lh * 8];
            const u16x8 ahi = *(const u16x8*)&Atile[cur][arow][16 + lh * 8];
            const u16x16 av = __builtin_shufflevector(alo, ahi,
                0, 1, 2, 3, 4, 5, 6, 7, 8, 9, 10, 11, 12, 13, 14, 15);
#pragma unroll
            for (int nt = 0; nt < 2; ++nt) {
                const int bcol = wn * 32 + nt * 16 + l16;
                // B fragment: bv[0..7] = K lh*8..+7 ; bv[8..15] = K 16+lh*8..+7 (transposed tile)
                const u16x8 blo = *(const u16x8*)&Btile[cur][bcol][lh * 8];
                const u16x8 bhi = *(const u16x8*)&Btile[cur][bcol][16 + lh * 8];
                const u16x16 bv = __builtin_shufflevector(blo, bhi,
                    0, 1, 2, 3, 4, 5, 6, 7, 8, 9, 10, 11, 12, 13, 14, 15);
                acc[mt][nt] = __builtin_amdgcn_wmma_f32_16x16x32_bf16(
                    false, __builtin_bit_cast(v16bf, av),
                    false, __builtin_bit_cast(v16bf, bv),
                    (short)0, acc[mt][nt], false, false);
            }
        }

        // ---- scatter next B tile (transposed) into the other buffer
        if (has_next) {
#pragma unroll
            for (int j = 0; j < 8; ++j) {
                Btile[nxt][bc + j][br]     = nb0[j];
                Btile[nxt][bc + 8 + j][br] = nb1[j];
            }
        }
        wait_asynccnt0();   // next A tile landed
        __syncthreads();    // everyone done reading cur; next buffer visible
    }

    // ---- store: VGPR v -> M = base + lh*8 + v, N = base + l16
#pragma unroll
    for (int mt = 0; mt < 4; ++mt) {
#pragma unroll
        for (int nt = 0; nt < 2; ++nt) {
            float* Cp = C + (size_t)(m0 + wm * 64 + mt * 16 + lh * 8) * N
                          + n0 + wn * 32 + nt * 16 + l16;
#pragma unroll
            for (int v = 0; v < 8; ++v)
                Cp[(size_t)v * N] = acc[mt][nt][v];
        }
    }
}

// ---------------- simple per-wave WMMA GEMM (for N=160 x_proj GEMM) --------
__global__ __launch_bounds__(256)
void wmma_gemm_bf16_kernel(const unsigned short* __restrict__ A, // MxK bf16
                           const unsigned short* __restrict__ B, // KxN bf16
                           float* __restrict__ C,                // MxN f32
                           int M, int N, int K) {
    const int wave  = (int)((blockIdx.x * blockDim.x + threadIdx.x) >> 5);
    const int lane  = threadIdx.x & 31;
    const int ntile = N >> 4;
    const int tm    = wave / ntile;
    const int tn    = wave - tm * ntile;
    if (tm >= (M >> 4)) return;            // whole-wave exit: EXEC all-ones at WMMA

    const int l16 = lane & 15;
    const int lh  = lane >> 4;
    const int arow = (tm << 4) + l16;
    const int bcol = (tn << 4) + l16;

    v8f acc = {};
    for (int k0 = 0; k0 < K; k0 += 32) {
        u16x16 av;
        const unsigned short* Ap = A + (size_t)arow * K + k0;
#pragma unroll
        for (int v = 0; v < 8; ++v) {
            const int kk = lh * 16 + v * 2;
            const unsigned int d = *(const unsigned int*)(Ap + kk);
            av[2 * v]     = (unsigned short)(d & 0xFFFFu);
            av[2 * v + 1] = (unsigned short)(d >> 16);
        }
        u16x16 bv;
        const unsigned short* Bp = B + (size_t)(k0 + lh * 16) * N + bcol;
#pragma unroll
        for (int v = 0; v < 8; ++v) {
            bv[2 * v]     = Bp[(size_t)(2 * v) * N];
            bv[2 * v + 1] = Bp[(size_t)(2 * v + 1) * N];
        }
        acc = __builtin_amdgcn_wmma_f32_16x16x32_bf16(
            false, __builtin_bit_cast(v16bf, av),
            false, __builtin_bit_cast(v16bf, bv),
            (short)0, acc, false, false);
    }
    float* Cp = C + (size_t)((tm << 4) + lh * 8) * N + (tn << 4) + l16;
#pragma unroll
    for (int v = 0; v < 8; ++v)
        Cp[(size_t)v * N] = acc[v];
}

// ---------------- depthwise causal conv (K=4) + bias + SiLU ----------------
__global__ void conv_silu_kernel(const float* __restrict__ proj,
                                 const float* __restrict__ w,   // (DI,1,4)
                                 const float* __restrict__ bias,
                                 float* __restrict__ xf,
                                 unsigned short* __restrict__ xb) {
    size_t i = (size_t)blockIdx.x * blockDim.x + threadIdx.x;
    if (i >= (size_t)BL * DI) return;
    const int d  = (int)(i % DI);
    const size_t bt = i / DI;
    const int t  = (int)(bt % LSEQ);
    const size_t base = bt - (size_t)t;  // b*LSEQ
    float acc = bias[d];
#pragma unroll
    for (int j = 0; j < KW; ++j) {
        const int tt = t - (KW - 1) + j;
        if (tt >= 0)
            acc += w[d * KW + j] * proj[(base + tt) * PROJW + d];
    }
    const float s = silu_f(acc);
    xf[i] = s;
    xb[i] = f32_to_bf16(s);
}

// ---------------- fused RMSNorm of dt(128) / B(16) / C(16), one wave per row
__global__ __launch_bounds__(256)
void rmsnorm_kernel(float* __restrict__ ssm,            // (BL,160) in-place
                    const float* __restrict__ dtw,
                    const float* __restrict__ bw,
                    const float* __restrict__ cw,
                    unsigned short* __restrict__ dt_bf) // (BL,128)
{
    const int row  = (int)((blockIdx.x * blockDim.x + threadIdx.x) >> 5);
    const int lane = threadIdx.x & 31;
    if (row >= BL) return;
    float* p = ssm + (size_t)row * SSMW;

    float v4[4];
    float s = 0.f;
#pragma unroll
    for (int j = 0; j < 4; ++j) { v4[j] = p[lane * 4 + j]; s += v4[j] * v4[j]; }
#pragma unroll
    for (int off = 16; off; off >>= 1) s += __shfl_xor(s, off, 32);
    const float r = rsqrtf(s * (1.0f / RK) + 1e-6f);
#pragma unroll
    for (int j = 0; j < 4; ++j) {
        const float o = v4[j] * r * dtw[lane * 4 + j];
        p[lane * 4 + j] = o;
        dt_bf[(size_t)row * RK + lane * 4 + j] = f32_to_bf16(o);
    }

    float vb = (lane < NS) ? p[RK + lane] : 0.f;
    float sb = vb * vb;
#pragma unroll
    for (int off = 16; off; off >>= 1) sb += __shfl_xor(sb, off, 32);
    const float rb = rsqrtf(sb * (1.0f / NS) + 1e-6f);
    if (lane < NS) p[RK + lane] = vb * rb * bw[lane];

    float vc = (lane < NS) ? p[RK + NS + lane] : 0.f;
    float sc = vc * vc;
#pragma unroll
    for (int off = 16; off; off >>= 1) sc += __shfl_xor(sc, off, 32);
    const float rc = rsqrtf(sc * (1.0f / NS) + 1e-6f);
    if (lane < NS) p[RK + NS + lane] = vc * rc * cw[lane];
}

// ---------------- selective scan: one thread per (b,d); h[16] in registers --
__global__ __launch_bounds__(256)
void scan_kernel(const float* __restrict__ deltapre, // (BL,DI)
                 const float* __restrict__ dtb,      // (DI,)
                 const float* __restrict__ ssm,      // (BL,160) normalized
                 const float* __restrict__ xconv,    // (BL,DI)
                 const float* __restrict__ proj,     // (BL,8192) gate at [4096+d]
                 const float* __restrict__ A_log,    // (DI,16)
                 const float* __restrict__ Dskip,    // (DI,)
                 unsigned short* __restrict__ y_bf)  // (BL,DI)
{
    const int d  = blockIdx.x * blockDim.x + threadIdx.x; // 0..DI-1
    const int bb = blockIdx.y;                            // 0..BSZ-1
    float A[NS], h[NS];
#pragma unroll
    for (int n = 0; n < NS; ++n) {
        A[n] = -__expf(A_log[(size_t)d * NS + n]);
        h[n] = 0.f;
    }
    const float Dv = Dskip[d];
    const float dbias = dtb[d];

    for (int t = 0; t < LSEQ; ++t) {
        const size_t rbt = (size_t)bb * LSEQ + t;
        if (t + 8 < LSEQ) {              // hide latency of the streaming reads
            const size_t rpf = rbt + 8;
            __builtin_prefetch(&deltapre[rpf * DI + d], 0, 0);
            __builtin_prefetch(&xconv[rpf * DI + d], 0, 0);
        }
        const float dp = deltapre[rbt * DI + d] + dbias;
        const float delta = (dp > 20.f) ? dp : log1pf(__expf(dp));
        const float u  = xconv[rbt * DI + d];
        const float du = delta * u;
        const float* Brow = ssm + rbt * SSMW + RK;
        const float* Crow = Brow + NS;
        float y = 0.f;
#pragma unroll
        for (int n = 0; n < NS; ++n) {
            const float hn = __expf(delta * A[n]) * h[n] + du * Brow[n];
            h[n] = hn;
            y += hn * Crow[n];
        }
        const float g = proj[rbt * PROJW + DI + d];
        const float out = (y + u * Dv) * silu_f(g);
        y_bf[rbt * DI + d] = f32_to_bf16(out);
    }
}

// ============================ host side ============================
extern "C" void kernel_launch(void* const* d_in, const int* in_sizes, int n_in,
                              void* d_out, int out_size, void* d_ws, size_t ws_size,
                              hipStream_t stream) {
    (void)in_sizes; (void)n_in; (void)out_size; (void)ws_size;
    const float* hs      = (const float*)d_in[0];
    const float* w_in    = (const float*)d_in[1];
    const float* conv_w  = (const float*)d_in[2];
    const float* conv_b  = (const float*)d_in[3];
    const float* w_xp    = (const float*)d_in[4];
    const float* dt_ln_w = (const float*)d_in[5];
    const float* b_ln_w  = (const float*)d_in[6];
    const float* c_ln_w  = (const float*)d_in[7];
    const float* w_dt    = (const float*)d_in[8];
    const float* dt_b    = (const float*)d_in[9];
    const float* A_log   = (const float*)d_in[10];
    const float* D_skip  = (const float*)d_in[11];
    const float* w_out   = (const float*)d_in[12];
    float* out = (float*)d_out;

    char* ws = (char*)d_ws;
    size_t off = 0;
    auto alloc = [&](size_t bytes) {
        off = (off + 255) & ~(size_t)255;
        size_t o = off; off += bytes; return o;
    };
    unsigned short* hs_bf   = (unsigned short*)(ws + alloc((size_t)BL * HDIM * 2));
    unsigned short* win_bf  = (unsigned short*)(ws + alloc((size_t)HDIM * PROJW * 2));
    float*          projf   = (float*)         (ws + alloc((size_t)BL * PROJW * 4));
    float*          xconvf  = (float*)         (ws + alloc((size_t)BL * DI * 4));
    unsigned short* xconvbf = (unsigned short*)(ws + alloc((size_t)BL * DI * 2));
    unsigned short* wxp_bf  = (unsigned short*)(ws + alloc((size_t)DI * SSMW * 2));
    float*          ssmf    = (float*)         (ws + alloc((size_t)BL * SSMW * 4));
    unsigned short* dt_bf   = (unsigned short*)(ws + alloc((size_t)BL * RK * 2));
    unsigned short* wdt_bf  = (unsigned short*)(ws + alloc((size_t)RK * DI * 2));
    float*          dpre    = (float*)         (ws + alloc((size_t)BL * DI * 4));
    unsigned short* y_bf    = (unsigned short*)(ws + alloc((size_t)BL * DI * 2));
    unsigned short* wout_bf = (unsigned short*)(ws + alloc((size_t)DI * HDIM * 2));

    auto cast_launch = [&](const float* src, unsigned short* dst, size_t n) {
        cast_f32_to_bf16_kernel<<<(unsigned)((n + 255) / 256), 256, 0, stream>>>(src, dst, n);
    };
    auto gemm_tiled = [&](const unsigned short* A, const unsigned short* B, float* C,
                          int M, int N, int K) {
        const int blocks = (M / BLK_M) * (N / BLK_N);
        wmma_gemm_bf16_tiled_kernel<<<blocks, 256, 0, stream>>>(A, B, C, M, N, K);
    };
    auto gemm_simple = [&](const unsigned short* A, const unsigned short* B, float* C,
                           int M, int N, int K) {
        const int waves  = (M >> 4) * (N >> 4);
        wmma_gemm_bf16_kernel<<<(waves + 7) / 8, 256, 0, stream>>>(A, B, C, M, N, K);
    };

    // 1) precision casts
    cast_launch(hs,    hs_bf,   (size_t)BL * HDIM);
    cast_launch(w_in,  win_bf,  (size_t)HDIM * PROJW);
    cast_launch(w_xp,  wxp_bf,  (size_t)DI * SSMW);
    cast_launch(w_dt,  wdt_bf,  (size_t)RK * DI);
    cast_launch(w_out, wout_bf, (size_t)DI * HDIM);

    // 2) proj = hs @ in_proj_w        (4096x2048)*(2048x8192)
    gemm_tiled(hs_bf, win_bf, projf, BL, PROJW, HDIM);

    // 3) depthwise conv + bias + SiLU
    {
        const size_t n = (size_t)BL * DI;
        conv_silu_kernel<<<(unsigned)((n + 255) / 256), 256, 0, stream>>>(
            projf, conv_w, conv_b, xconvf, xconvbf);
    }

    // 4) ssm = x @ x_proj_w           (4096x4096)*(4096x160): N=160 -> simple kernel
    gemm_simple(xconvbf, wxp_bf, ssmf, BL, SSMW, DI);

    // 5) RMSNorm dt/B/C
    rmsnorm_kernel<<<(BL * 32 + 255) / 256, 256, 0, stream>>>(
        ssmf, dt_ln_w, b_ln_w, c_ln_w, dt_bf);

    // 6) deltapre = dt @ dt_proj_w    (4096x128)*(128x4096)
    gemm_tiled(dt_bf, wdt_bf, dpre, BL, DI, RK);

    // 7) selective scan, fused epilogue
    {
        dim3 grid(DI / 256, BSZ);
        scan_kernel<<<grid, 256, 0, stream>>>(
            dpre, dt_b, ssmf, xconvf, projf, A_log, D_skip, y_bf);
    }

    // 8) out = y @ out_proj_w         (4096x4096)*(4096x2048)
    gemm_tiled(y_bf, wout_bf, out, BL, HDIM, DI);
}
